// CellBox_66151086293486
// MI455X (gfx1250) — matrix operand order repaired and checked by reading, not compiled, not run on hardware
//
#include <hip/hip_runtime.h>
#include <hip/hip_bf16.h>
#include <math.h>

#define N_X        1024
#define BATCH      1024
#define N_PROTEIN  820
#define N_ACTIVITY 900
#define DT         0.1f
#define N_T        100
#define LAST_STEPS 5

typedef __bf16 v16bf __attribute__((ext_vector_type(16)));
typedef __bf16 v8bf  __attribute__((ext_vector_type(8)));
typedef float  v8f   __attribute__((ext_vector_type(8)));

__device__ __forceinline__ float softplus_f(float x) {
    return (x > 20.f) ? x : log1pf(expf(x));
}

// CDNA5 has a hardware V_TANH_F32 transcendental; use it if the builtin
// exists on this toolchain, otherwise fall back to libm tanhf.
__device__ __forceinline__ float fast_tanhf(float x) {
#if __has_builtin(__builtin_amdgcn_tanhf)
    return __builtin_amdgcn_tanhf(x);
#elif __has_builtin(__builtin_amdgcn_tanh_f32)
    return __builtin_amdgcn_tanh_f32(x);
#else
    return tanhf(x);
#endif
}

// Tiled prep: one 32x32 tile per block (1024 blocks, 256 threads, 4 elems/thread).
// Builds masked bf16 W, zeroes accumulators, seeds state from y0, and does the
// mu / y0 transposes through padded LDS tiles so all global accesses coalesce.
__global__ __launch_bounds__(256) void prep_kernel(
    const float* __restrict__ y0, const float* __restrict__ mu,
    const float* __restrict__ W, const float* __restrict__ alpha_raw,
    const float* __restrict__ eps_raw,
    __bf16* __restrict__ Wm, float* __restrict__ mu_t,
    float* __restrict__ xf, __bf16* __restrict__ xT,
    float* __restrict__ sum, float* __restrict__ sumsq,
    float* __restrict__ alpha, float* __restrict__ eps)
{
    __shared__ float tmu[32][33];   // tmu[r][c] = mu[j0+r][i0+c]
    __shared__ float tx[32][33];    // tx[r][c]  = y0[i0+r][j0+c]

    const int c  = threadIdx.x & 31;
    const int r4 = threadIdx.x >> 5;          // 0..7
    const int i0 = (blockIdx.x & 31) * 32;    // x-index tile origin
    const int j0 = (blockIdx.x >> 5) * 32;    // col/batch tile origin

#pragma unroll
    for (int rr = 0; rr < 4; ++rr) {
        const int r = r4 * 4 + rr;
        const int i = i0 + r;
        const int j = j0 + c;
        const size_t idx = (size_t)i * N_X + j;

        float m = (i == j) ? 0.f : 1.f;
        if (i >= N_ACTIVITY) m = 0.f;
        if (j >= N_PROTEIN && j < N_ACTIVITY) m = 0.f;
        if (i >= N_PROTEIN && i < N_ACTIVITY && j >= N_ACTIVITY) m = 0.f;
        Wm[idx] = (__bf16)(m * W[idx]);

        sum[idx] = 0.f;
        sumsq[idx] = 0.f;

        const float x0 = y0[idx];
        xf[idx] = x0;
        tx[r][c] = x0;
        tmu[r][c] = mu[(size_t)(j0 + r) * N_X + (i0 + c)];  // coalesced mu read
    }
    __syncthreads();
#pragma unroll
    for (int rr = 0; rr < 4; ++rr) {
        const int r = r4 * 4 + rr;
        // mu_t[i,b] = mu[b,i]:  i = i0+r, b = j0+c  ->  tmu[c][r]
        mu_t[(size_t)(i0 + r) * BATCH + (j0 + c)] = tmu[c][r];
        // xT[b,i] = y0[i,b]:    b = j0+r, i = i0+c  ->  tx[c][r]
        xT[(size_t)(j0 + r) * N_X + (i0 + c)] = (__bf16)tx[c][r];
    }

    const int tid = blockIdx.x * 256 + threadIdx.x;
    if (tid < N_X) {
        alpha[tid] = softplus_f(alpha_raw[tid]);
        eps[tid]   = softplus_f(eps_raw[tid]);
    }
}

// Fused GEMM (Wm @ x via bf16 WMMA, f32 accumulate) + Euler update.
// mode 0: step, mode 1: step + accumulate sum/sumsq, mode 2: emit dx only.
// Each wave computes a 32x32 output block = 2x2 WMMA 16x16 tiles.
__global__ __launch_bounds__(256) void step_kernel(
    const __bf16* __restrict__ Wm, const __bf16* __restrict__ xT,
    const float* __restrict__ xcur, const float* __restrict__ mu_t,
    const float* __restrict__ alpha, const float* __restrict__ eps,
    float* __restrict__ xnext, __bf16* __restrict__ xTnext,
    float* __restrict__ sum, float* __restrict__ sumsq,
    float* __restrict__ dxout, int mode)
{
    const int lane = threadIdx.x & 31;
    const int wave = threadIdx.x >> 5;
    const int tile = blockIdx.x * 8 + wave;   // 0..1023 (32x32 grid of blocks)
    const int m0 = (tile >> 5) << 5;
    const int n0 = (tile & 31) << 5;

    const int half = lane >> 4;   // K-half selector per ISA A/B lane layout
    const int l16  = lane & 15;

    v8f c00 = {}, c01 = {}, c10 = {}, c11 = {};

    // A (row-major Wm): lane holds row m, K run of 16 at half*16.
    const __bf16* arow0 = Wm + (size_t)(m0 + l16) * N_X + half * 16;
    const __bf16* arow1 = arow0 + (size_t)16 * N_X;
    // B (batch-major xT): lane holds column b = n+l16, K run of 16 at half*16.
    const __bf16* brow0 = xT + (size_t)(n0 + l16) * N_X + half * 16;
    const __bf16* brow1 = brow0 + (size_t)16 * N_X;

    for (int k = 0; k < N_X; k += 32) {
        v8bf a0lo = *(const v8bf*)(arow0 + k);
        v8bf a0hi = *(const v8bf*)(arow0 + k + 8);
        v8bf a1lo = *(const v8bf*)(arow1 + k);
        v8bf a1hi = *(const v8bf*)(arow1 + k + 8);
        v16bf b0  = *(const v16bf*)(brow0 + k);
        v16bf b1  = *(const v16bf*)(brow1 + k);
        v16bf a0, a1;
#pragma unroll
        for (int e = 0; e < 8; ++e) {
            a0[e] = a0lo[e]; a0[8 + e] = a0hi[e];
            a1[e] = a1lo[e]; a1[8 + e] = a1hi[e];
        }
        c00 = __builtin_amdgcn_wmma_f32_16x16x32_bf16(false, a0, false, b0, (short)0, c00, false, false);
        c01 = __builtin_amdgcn_wmma_f32_16x16x32_bf16(false, a0, false, b1, (short)0, c01, false, false);
        c10 = __builtin_amdgcn_wmma_f32_16x16x32_bf16(false, a1, false, b0, (short)0, c10, false, false);
        c11 = __builtin_amdgcn_wmma_f32_16x16x32_bf16(false, a1, false, b1, (short)0, c11, false, false);
    }

    // Epilogue: C tile layout — VGPR r holds (m = mt + r + half*8, n = nt + l16).
    auto epilogue = [&](int mt, int nt, const v8f& c) {
#pragma unroll
        for (int r = 0; r < 8; ++r) {
            const int i = mt + r + half * 8;
            const int b = nt + l16;
            const size_t idx = (size_t)i * BATCH + b;
            const float z  = c[r] + mu_t[idx];
            const float t  = fast_tanhf(z);
            const float xo = xcur[idx];
            const float dx = eps[i] * t - alpha[i] * xo;
            if (mode == 2) {
                dxout[idx] = dx;
            } else {
                const float xn = xo + DT * dx;
                xnext[idx] = xn;
                xTnext[(size_t)b * N_X + i] = (__bf16)xn;
                if (mode == 1) { sum[idx] += xn; sumsq[idx] += xn * xn; }
            }
        }
    };
    epilogue(m0,      n0,      c00);
    epilogue(m0,      n0 + 16, c01);
    epilogue(m0 + 16, n0,      c10);
    epilogue(m0 + 16, n0 + 16, c11);
}

// mean/sd (ddof=1) from online sums; yhat = x_final^T via padded LDS tile
// so both the read and the transposed write are lane-contiguous.
__global__ __launch_bounds__(256) void fin_kernel(
    const float* __restrict__ sum, const float* __restrict__ sumsq,
    const float* __restrict__ xfin, float* __restrict__ out)
{
    __shared__ float tile[32][33];
    const size_t M = (size_t)N_X * BATCH;

    const int c  = threadIdx.x & 31;
    const int r4 = threadIdx.x >> 5;
    const int i0 = (blockIdx.x & 31) * 32;
    const int b0 = (blockIdx.x >> 5) * 32;

#pragma unroll
    for (int rr = 0; rr < 4; ++rr) {
        const int r = r4 * 4 + rr;
        const size_t idx = (size_t)(i0 + r) * BATCH + (b0 + c);
        const float s = sum[idx];
        const float q = sumsq[idx];
        const float mean = s * (1.0f / LAST_STEPS);
        const float var  = (q - s * mean) * (1.0f / (LAST_STEPS - 1));
        out[idx]     = mean;
        out[M + idx] = sqrtf(fmaxf(var, 0.f));
        tile[r][c] = xfin[idx];
    }
    __syncthreads();
#pragma unroll
    for (int rr = 0; rr < 4; ++rr) {
        const int r = r4 * 4 + rr;
        // yhat[b,i] = x[i,b]:  b = b0+r, i = i0+c  ->  tile[c][r]
        out[3 * M + (size_t)(b0 + r) * N_X + (i0 + c)] = tile[c][r];
    }
}

extern "C" void kernel_launch(void* const* d_in, const int* in_sizes, int n_in,
                              void* d_out, int out_size, void* d_ws, size_t ws_size,
                              hipStream_t stream) {
    const float* y0 = (const float*)d_in[0];
    const float* mu = (const float*)d_in[1];
    const float* W  = (const float*)d_in[2];
    const float* ar = (const float*)d_in[3];
    const float* er = (const float*)d_in[4];
    float* out = (float*)d_out;

    const size_t M = (size_t)N_X * BATCH;
    char* p = (char*)d_ws;
    __bf16* Wm   = (__bf16*)p; p += M * 2;
    float*  mu_t = (float*)p;  p += M * 4;
    float*  xf0  = (float*)p;  p += M * 4;
    float*  xf1  = (float*)p;  p += M * 4;
    __bf16* xT0  = (__bf16*)p; p += M * 2;
    __bf16* xT1  = (__bf16*)p; p += M * 2;
    float*  sum  = (float*)p;  p += M * 4;
    float*  sumq = (float*)p;  p += M * 4;
    float*  alph = (float*)p;  p += N_X * 4;
    float*  eps  = (float*)p;  p += N_X * 4;

    float*  xf[2] = { xf0, xf1 };
    __bf16* xT[2] = { xT0, xT1 };

    prep_kernel<<<1024, 256, 0, stream>>>(
        y0, mu, W, ar, er, Wm, mu_t, xf[0], xT[0], sum, sumq, alph, eps);

    for (int t = 0; t < N_T; ++t) {
        const int cur = t & 1, nxt = cur ^ 1;
        const int mode = (t >= N_T - LAST_STEPS) ? 1 : 0;
        step_kernel<<<128, 256, 0, stream>>>(
            Wm, xT[cur], xf[cur], mu_t, alph, eps,
            xf[nxt], xT[nxt], sum, sumq, (float*)nullptr, mode);
    }

    // N_T is even -> final state lives in slot 0. dx_final into out[2M..3M).
    step_kernel<<<128, 256, 0, stream>>>(
        Wm, xT[0], xf[0], mu_t, alph, eps,
        (float*)nullptr, (__bf16*)nullptr, (float*)nullptr, (float*)nullptr,
        out + 2 * M, 2);

    fin_kernel<<<1024, 256, 0, stream>>>(sum, sumq, xf[0], out);
}
